// LlamaAttentionVerify_77790447665855
// MI455X (gfx1250) — matrix operand order repaired and checked by reading, not compiled, not run on hardware
//
#include <hip/hip_runtime.h>

typedef __attribute__((ext_vector_type(16))) _Float16 v16h;
typedef __attribute__((ext_vector_type(8)))  float    v8f;
typedef int v4i __attribute__((vector_size(16)));

constexpr int S_LEN  = 2048;
constexpr int HIDN   = 4096;
constexpr int NHEAD  = 32;
constexpr int NKVH   = 8;
constexpr int HDIM   = 128;
constexpr int QKV_N  = 6144;   // 4096 Q + 1024 K + 1024 V
constexpr int K_OFF  = 4096;
constexpr int V_OFF  = 5120;
constexpr float SCALING = 0.08838834764831845f; // 128^-0.5

// ---------------- async global->LDS (CDNA5), guarded ----------------
// Probe-confirmed (round 2 stderr): builtin exists, param0 = v4i addrspace(1)*,
// 4 args total (gaddr, lds, offset, cpol).
#if __has_builtin(__builtin_amdgcn_global_load_async_to_lds_b128)
#define HAVE_ASYNC 1
#define ASYNC_COPY16(gsrc, ldst)                                        \
  __builtin_amdgcn_global_load_async_to_lds_b128(                       \
      (__attribute__((address_space(1))) v4i*)(gsrc),                   \
      (__attribute__((address_space(3))) v4i*)(ldst), 0, 0)
#define ASYNC_WAIT() asm volatile("s_wait_asynccnt 0" ::: "memory")
#else
#define HAVE_ASYNC 0
#define ASYNC_WAIT() ((void)0)
#endif

union HU16 { uint4 u; _Float16 h[8]; };

// ---------------- WMMA helpers (gfx1250, wave32) ----------------

static __device__ __forceinline__ v8f wmma_f16(v16h a, v16h b, v8f c) {
  // D = A(16x32 f16) * B(32x16 f16) + C(16x16 f32)
  return __builtin_amdgcn_wmma_f32_16x16x32_f16(
      false, a, false, b, (short)0, c, false, false);
}

// A-matrix 16x32 f16 fragment from row-major LDS tile [16][stride].
// Lane l(0..15): row=l, k = {kb..kb+15}, kb = (l>=16)?16:0.
static __device__ __forceinline__ v16h frag_a(const _Float16* base, int stride, int lane) {
  const int row = lane & 15;
  const int kb  = (lane >> 4) << 4;
  const _Float16* p = base + row * stride;
  v16h a;
#pragma unroll
  for (int i = 0; i < 8; ++i) { a[i] = p[kb + i]; a[8 + i] = p[kb + 8 + i]; }
  return a;
}

// B-matrix 32x16 f16 fragment from LDS stored [K][N] (row stride = stride).
static __device__ __forceinline__ v16h frag_b(const _Float16* base, int stride, int lane) {
  const int col  = lane & 15;
  const int koff = (lane >> 4) << 4;
  v16h b;
#pragma unroll
  for (int i = 0; i < 16; ++i) b[i] = base[(koff + i) * stride + col];
  return b;
}

// B fragment where B = T^T, T stored row-major [N][K] (stride) -> contiguous reads.
static __device__ __forceinline__ v16h frag_bT(const _Float16* base, int stride, int lane) {
  const int col  = lane & 15;
  const int koff = (lane >> 4) << 4;
  const _Float16* p = base + col * stride + koff;
  v16h b;
#pragma unroll
  for (int i = 0; i < 16; ++i) b[i] = p[i];
  return b;
}

// ---------------- elementwise kernels ----------------

__global__ void f32_to_f16_kernel(const float* __restrict__ src,
                                  _Float16* __restrict__ dst, int n) {
  int i = blockIdx.x * blockDim.x + threadIdx.x;
  if (i < n) dst[i] = (_Float16)src[i];
}

// In-place RoPE on f16 QKV buffer. 40 "heads" (32 Q + 8 K), 64 (d,d+64) pairs.
__global__ void rope_kernel(_Float16* __restrict__ qkv,
                            const float* __restrict__ cosp,
                            const float* __restrict__ sinp) {
  int idx = blockIdx.x * blockDim.x + threadIdx.x;
  int d  = idx & 63;
  int hh = (idx >> 6) % (NHEAD + NKVH);
  int s  = idx / (64 * (NHEAD + NKVH));
  if (s >= S_LEN) return;
  int base = (hh < NHEAD) ? hh * HDIM : K_OFF + (hh - NHEAD) * HDIM;
  _Float16* p = qkv + (size_t)s * QKV_N + base;
  float x1 = (float)p[d], x2 = (float)p[d + 64];
  float c = cosp[s * HDIM + d], sn = sinp[s * HDIM + d]; // second halves duplicate
  p[d]      = (_Float16)(x1 * c - x2 * sn);
  p[d + 64] = (_Float16)(x2 * c + x1 * sn);
}

// ---------------- tiled GEMM: C[M,N] = A[M,K](f16) @ W[N,K](f32)^T ----------------
// BM=128, BN=64, BK=32, 256 threads = 8 waves (4x2), double-buffered LDS.

constexpr int BM = 128, BN = 64, BK = 32;
constexpr int AST = BK + 8;   // 40 halfs, keeps 16B alignment per 16-half chunk
constexpr int BST = BN + 8;   // 72 halfs

template <typename OutT>
__global__ __launch_bounds__(256) void gemm_abt_kernel(
    const _Float16* __restrict__ A, const float* __restrict__ W,
    OutT* __restrict__ C, int M, int N, int K, int ldc) {
  __shared__ _Float16 As[2][BM * AST];
  __shared__ _Float16 Bs[2][BK * BST];
  (void)M; (void)N; // grid exactly tiles M,N

  const int tid  = threadIdx.x;
  const int lane = tid & 31;
  const int w    = tid >> 5;
  const int wm   = (w & 3) * 32;   // wave row offset in tile
  const int wn   = (w >> 2) * 32;  // wave col offset in tile
  const int bm   = blockIdx.y * BM;
  const int bn   = blockIdx.x * BN;

  v8f vzero = {};
  v8f acc[2][2];
#pragma unroll
  for (int i = 0; i < 2; ++i)
#pragma unroll
    for (int j = 0; j < 2; ++j) acc[i][j] = vzero;

  auto load_tiles = [&](int buf, int k0) {
    // A tile: 128x32 halfs, 2 x 16B per thread (async direct-to-LDS if available)
    {
      int m  = tid >> 1;
      int kk = (tid & 1) * 16;
      const _Float16* src = A + (size_t)(bm + m) * K + k0 + kk;
      _Float16* dst = &As[buf][m * AST + kk];
#if HAVE_ASYNC
      ASYNC_COPY16(src, dst);
      ASYNC_COPY16(src + 8, dst + 8);
#else
      reinterpret_cast<uint4*>(dst)[0] = reinterpret_cast<const uint4*>(src)[0];
      reinterpret_cast<uint4*>(dst)[1] = reinterpret_cast<const uint4*>(src)[1];
#endif
    }
    // B tile: convert f32 weights -> f16, transpose into [K][N]
    {
      int n  = tid >> 2;
      int kc = (tid & 3) * 8;
      const float* src = W + (size_t)(bn + n) * K + k0 + kc;
      __builtin_prefetch(src + BK, 0, 1); // global_prefetch next k-slab
#pragma unroll
      for (int i = 0; i < 8; ++i)
        Bs[buf][(kc + i) * BST + n] = (_Float16)src[i];
    }
  };

  const int nkb = K / BK;
  load_tiles(0, 0);
  ASYNC_WAIT();
  __syncthreads();

  for (int kb = 0; kb < nkb; ++kb) {
    const int cur = kb & 1;
    if (kb + 1 < nkb) load_tiles(cur ^ 1, (kb + 1) * BK);

    v16h af0 = frag_a(&As[cur][(wm +  0) * AST], AST, lane);
    v16h af1 = frag_a(&As[cur][(wm + 16) * AST], AST, lane);
    v16h bf0 = frag_b(&Bs[cur][wn +  0], BST, lane);
    v16h bf1 = frag_b(&Bs[cur][wn + 16], BST, lane);
    acc[0][0] = wmma_f16(af0, bf0, acc[0][0]);
    acc[0][1] = wmma_f16(af0, bf1, acc[0][1]);
    acc[1][0] = wmma_f16(af1, bf0, acc[1][0]);
    acc[1][1] = wmma_f16(af1, bf1, acc[1][1]);

    ASYNC_WAIT(); // next-tile async copies complete; overlapped with the WMMAs above
    __syncthreads();
  }

  // C/D layout: lane holds col = lane&15, rows r + 8*(lane>>4)
  const int r0 = (lane >> 4) * 8;
  const int cc = lane & 15;
#pragma unroll
  for (int mf = 0; mf < 2; ++mf)
#pragma unroll
    for (int nf = 0; nf < 2; ++nf)
#pragma unroll
      for (int r = 0; r < 8; ++r) {
        int row = bm + wm + mf * 16 + r0 + r;
        int col = bn + wn + nf * 16 + cc;
        C[(size_t)row * ldc + col] = (OutT)acc[mf][nf][r];
      }
}

// ---------------- flash attention ----------------
// Block = (q-block of 64 rows, head). 128 threads = 4 waves x 16 q-rows.

constexpr int AQ = 64, AK = 32;
constexpr int KST = HDIM + 8;  // 136 halfs (K tile row stride)
constexpr int VTS = AK + 8;    // 40 halfs (V^T tile row stride)
constexpr int PST = AK + 8;    // 40 halfs

__global__ __launch_bounds__(128) void attn_kernel(
    const _Float16* __restrict__ qkv, _Float16* __restrict__ ctx) {
  __shared__ _Float16 Ks[AK * KST];     // K tile, row-major [kv][d]
  __shared__ _Float16 Vt[HDIM * VTS];   // V tile, transposed [d][kv]
  __shared__ _Float16 Ps[4][16 * PST];

  const int tid  = threadIdx.x;
  const int lane = tid & 31;
  const int w    = tid >> 5;              // 0..3
  const int h    = blockIdx.y;
  const int q0   = blockIdx.x * AQ;
  const int kvh  = h >> 2;                // N_REP = 4
  const int kb8  = (lane >> 4) << 3;      // A-frag k-base (0|8)
  const int r0   = (lane >> 4) * 8;
  const int cc   = lane & 15;

  // Q fragments (A layout), 16x128 per wave as 4 frags of 16x32; 2x b128 each
  const int qrow = q0 + w * 16 + (lane & 15);
  const _Float16* qptr = qkv + (size_t)qrow * QKV_N + h * HDIM;
  v16h qf[4];
#pragma unroll
  for (int kk = 0; kk < 4; ++kk) {
    HU16 u0, u1;
    u0.u = *reinterpret_cast<const uint4*>(qptr + 32 * kk + ((lane >> 4) << 4));
    u1.u = *reinterpret_cast<const uint4*>(qptr + 32 * kk + ((lane >> 4) << 4) + 8);
#pragma unroll
    for (int i = 0; i < 8; ++i) { qf[kk][i] = u0.h[i]; qf[kk][8 + i] = u1.h[i]; }
  }

  v8f vzero = {};
  float m_i[8], l_i[8];
  v8f o[8];
#pragma unroll
  for (int r = 0; r < 8; ++r) { m_i[r] = -1e30f; l_i[r] = 0.0f; }
#pragma unroll
  for (int nf = 0; nf < 8; ++nf) o[nf] = vzero;

  const int rowhi = q0 + w * 16 + 15;
  const int nkb = (q0 + AQ) / AK;

  for (int kb = 0; kb < nkb; ++kb) {
    { // cooperative tile staging: 32 kv rows x 128 halfs for both K and V
      int kr = tid & 31;
      int dc = (tid >> 5) * 32;
      const _Float16* kgs = qkv + (size_t)(kb * AK + kr) * QKV_N + K_OFF + kvh * HDIM + dc;
      const _Float16* vgs = qkv + (size_t)(kb * AK + kr) * QKV_N + V_OFF + kvh * HDIM + dc;
      // K: straight copy (async direct-to-LDS when available)
#pragma unroll
      for (int i = 0; i < 4; ++i) {
#if HAVE_ASYNC
        ASYNC_COPY16(kgs + i * 8, &Ks[kr * KST + dc + i * 8]);
#else
        reinterpret_cast<uint4*>(&Ks[kr * KST + dc])[i] =
            reinterpret_cast<const uint4*>(kgs)[i];
#endif
      }
      // V: transpose into Vt[d][kv] so P*V fragments are contiguous b128 reads
#pragma unroll
      for (int c = 0; c < 4; ++c) {
        HU16 vv;
        vv.u = *reinterpret_cast<const uint4*>(vgs + c * 8);
#pragma unroll
        for (int i = 0; i < 8; ++i)
          Vt[(dc + c * 8 + i) * VTS + kr] = vv.h[i];
      }
    }
    ASYNC_WAIT();
    __syncthreads();

    if (kb * AK <= rowhi) { // wave-uniform: skip fully-masked tiles
      // S = Q * K^T  (16x32 per wave, 2 n-frags, 4 k-steps)
      v8f sacc0 = vzero, sacc1 = vzero;
#pragma unroll
      for (int kk = 0; kk < 4; ++kk) {
        sacc0 = wmma_f16(qf[kk], frag_bT(&Ks[0 * 16 * KST + kk * 32], KST, lane), sacc0);
        sacc1 = wmma_f16(qf[kk], frag_bT(&Ks[1 * 16 * KST + kk * 32], KST, lane), sacc1);
      }
      // scale + causal mask (sentinel -1e30; kv block 0 seen first, key 0 valid for every row -> no NaN)
      float s0[8], s1[8], alpha[8];
#pragma unroll
      for (int r = 0; r < 8; ++r) {
        int row = q0 + w * 16 + r0 + r;
        s0[r] = (kb * AK + cc      <= row) ? sacc0[r] * SCALING : -1e30f;
        s1[r] = (kb * AK + 16 + cc <= row) ? sacc1[r] * SCALING : -1e30f;
      }
      // online softmax: each q-row lives on the 16 lanes of one half-wave
#pragma unroll
      for (int r = 0; r < 8; ++r) {
        float rm = fmaxf(s0[r], s1[r]);
#pragma unroll
        for (int off = 8; off >= 1; off >>= 1)
          rm = fmaxf(rm, __shfl_xor(rm, off, 32));
        float mn = fmaxf(m_i[r], rm);
        alpha[r] = __expf(m_i[r] - mn);
        float p0 = __expf(s0[r] - mn);
        float p1 = __expf(s1[r] - mn);
        float ps = p0 + p1;
#pragma unroll
        for (int off = 8; off >= 1; off >>= 1)
          ps += __shfl_xor(ps, off, 32);
        l_i[r] = alpha[r] * l_i[r] + ps;
        m_i[r] = mn;
        Ps[w][(r0 + r) * PST + cc]      = (_Float16)p0;
        Ps[w][(r0 + r) * PST + 16 + cc] = (_Float16)p1;
      }
      // wave-private LDS transpose: DS ops are in-order per wave; block compiler reorder
      asm volatile("s_wait_dscnt 0" ::: "memory");
      v16h pf;
      {
        const _Float16* pp = &Ps[w][(lane & 15) * PST];
#pragma unroll
        for (int i = 0; i < 8; ++i) {
          pf[i]     = pp[((lane >> 4) << 4) + i];
          pf[8 + i] = pp[((lane >> 4) << 4) + 8 + i];
        }
      }
      // O = diag(alpha)*O + P*V  (8 n-frags of 16 cols; V^T rows are contiguous)
      const int koff = (lane >> 4) << 4;
#pragma unroll
      for (int nf = 0; nf < 8; ++nf) {
#pragma unroll
        for (int r = 0; r < 8; ++r) o[nf][r] *= alpha[r];
        const _Float16* vp = &Vt[(nf * 16 + cc) * VTS + koff];
        v16h vf;
#pragma unroll
        for (int i = 0; i < 16; ++i) vf[i] = vp[i];
        o[nf] = wmma_f16(pf, vf, o[nf]);
      }
    }
    __syncthreads();
  }

  // normalize + store ctx [s][h*128+d] as f16
#pragma unroll
  for (int r = 0; r < 8; ++r) {
    float inv = 1.0f / l_i[r];
    int row = q0 + w * 16 + r0 + r;
#pragma unroll
    for (int nf = 0; nf < 8; ++nf)
      ctx[(size_t)row * HIDN + h * HDIM + nf * 16 + cc] = (_Float16)(o[nf][r] * inv);
  }
}

// ---------------- launch ----------------

extern "C" void kernel_launch(void* const* d_in, const int* in_sizes, int n_in,
                              void* d_out, int out_size, void* d_ws, size_t ws_size,
                              hipStream_t stream) {
  (void)in_sizes; (void)n_in; (void)out_size; (void)ws_size;
  const float* hidden = (const float*)d_in[0];
  const float* cosp   = (const float*)d_in[1];
  const float* sinp   = (const float*)d_in[2];
  // d_in[3] attention_mask: pure causal, handled analytically
  const float* q_w    = (const float*)d_in[4];
  const float* k_w    = (const float*)d_in[5];
  const float* v_w    = (const float*)d_in[6];
  const float* o_w    = (const float*)d_in[7];
  float* out = (float*)d_out;

  char* ws = (char*)d_ws;
  _Float16* hs  = (_Float16*)ws;                                  // 16 MB
  _Float16* qkv = (_Float16*)(ws + (size_t)(16 << 20));           // 24 MB
  _Float16* ctx = (_Float16*)(ws + (size_t)(16 << 20) + (size_t)(24 << 20)); // 16 MB

  // 1) hidden f32 -> f16
  {
    int n = S_LEN * HIDN;
    f32_to_f16_kernel<<<(n + 255) / 256, 256, 0, stream>>>(hidden, hs, n);
  }
  // 2-4) fused QKV projection into qkv[s][6144]
  gemm_abt_kernel<_Float16><<<dim3(HIDN / BN, S_LEN / BM), 256, 0, stream>>>(
      hs, q_w, qkv, S_LEN, HIDN, HIDN, QKV_N);
  gemm_abt_kernel<_Float16><<<dim3((NKVH * HDIM) / BN, S_LEN / BM), 256, 0, stream>>>(
      hs, k_w, qkv + K_OFF, S_LEN, NKVH * HDIM, HIDN, QKV_N);
  gemm_abt_kernel<_Float16><<<dim3((NKVH * HDIM) / BN, S_LEN / BM), 256, 0, stream>>>(
      hs, v_w, qkv + V_OFF, S_LEN, NKVH * HDIM, HIDN, QKV_N);
  // 5) RoPE in place on Q and K
  {
    int n = S_LEN * (NHEAD + NKVH) * 64;
    rope_kernel<<<(n + 255) / 256, 256, 0, stream>>>(qkv, cosp, sinp);
  }
  // 6) flash attention -> ctx f16 [s][4096]
  attn_kernel<<<dim3(S_LEN / AQ, NHEAD), 128, 0, stream>>>(qkv, ctx);
  // 7) output projection -> f32 out
  gemm_abt_kernel<float><<<dim3(HIDN / BN, S_LEN / BM), 256, 0, stream>>>(
      ctx, o_w, out, S_LEN, HIDN, HIDN, HIDN);
}
